// PixelPropagationModule_47571057771204
// MI455X (gfx1250) — matrix operand order repaired and checked by reading, not compiled, not run on hardware
//
#include <hip/hip_runtime.h>
#include <hip/hip_bf16.h>

typedef __bf16 bf16_t;
typedef __attribute__((ext_vector_type(16))) __bf16 v16bf;
typedef __attribute__((ext_vector_type(8)))  __bf16 v8bf;
typedef __attribute__((ext_vector_type(8)))  float  v8f;

#define NB   32
#define C    256
#define HW   2304
#define NPB  (C*HW)          // 589824 elements per batch
#define STAT_CNT (NB*HW)     // BN reduction count = 73728
#define BN_EPS 1e-5f

// ---------------------------------------------------------------------------
// Kernel 1: per-batch inverse squared L2 norm  inv2[b] = 1/||x_b||^2
// ---------------------------------------------------------------------------
__global__ __launch_bounds__(256)
void k_norm(const float* __restrict__ x, float* __restrict__ inv2) {
  __shared__ float red[256];
  const int b = blockIdx.x, tid = threadIdx.x;
  const float* xb = x + (size_t)b * NPB;
  float s = 0.f;
  for (int i = tid; i < NPB; i += 256) { float v = xb[i]; s += v * v; }
  red[tid] = s; __syncthreads();
  for (int st = 128; st > 0; st >>= 1) {
    if (tid < st) red[tid] += red[tid + st];
    __syncthreads();
  }
  if (tid == 0) inv2[b] = 1.0f / red[0];
}

// ---------------------------------------------------------------------------
// Kernel 2: XT[b][p][c] = bf16(x[b][c][p])  (tiled transpose, f32 -> bf16)
// Serves as K-contiguous B-operand storage for BOTH the cos matmul and conv.
// ---------------------------------------------------------------------------
__global__ __launch_bounds__(256)
void k_transpose_x(const float* __restrict__ x, bf16_t* __restrict__ xt) {
  __shared__ float tile[32][33];
  const int m0 = blockIdx.x * 32, k0 = blockIdx.y * 32, b = blockIdx.z;
  const int tx = threadIdx.x, ty = threadIdx.y;      // 32 x 8
  const size_t xb = (size_t)b * NPB;
#pragma unroll
  for (int r = 0; r < 4; ++r)
    tile[ty + r * 8][tx] = x[xb + (size_t)(k0 + ty + r * 8) * HW + m0 + tx];
  __syncthreads();
#pragma unroll
  for (int r = 0; r < 4; ++r)
    xt[xb + (size_t)(m0 + ty + r * 8) * C + k0 + tx] = (bf16_t)tile[tx][ty + r * 8];
}

// ---------------------------------------------------------------------------
// Kernel 2b: pre-swizzle conv weights into per-lane A-fragment order:
//   wfrag[((ot*8+kc)*32 + lane)*16 + e] = bf16(W[o][c])
//   o = ot*16 + lane%16 ; c = kc*32 + 16*(lane/16) + e
// 128 KB total, L2-resident; conv hot loop becomes pure v16bf load + WMMA.
// ---------------------------------------------------------------------------
__global__ __launch_bounds__(256)
void k_wcast(const float* __restrict__ w, bf16_t* __restrict__ wfrag) {
  const int idx  = blockIdx.x * 256 + threadIdx.x;   // 65536 elements
  const int e    = idx & 15;
  const int lane = (idx >> 4) & 31;
  const int kc   = (idx >> 9) & 7;
  const int ot   = idx >> 12;
  const int o = ot * 16 + (lane & 15);
  const int c = kc * 32 + 16 * (lane >> 4) + e;
  wfrag[idx] = (bf16_t)w[(size_t)o * C + c];
}

// ---------------------------------------------------------------------------
// Kernel 3: 1x1 conv via bf16 WMMA:  t[o][p] = sum_c W[o][c] x[c][p] + bias[o]
// Stores raw t (bf16). BN stats: raw values staged in LDS [256 ch][128 slots]
// (128 KB dynamic LDS, WGP has 320 KB), then one cooperative reduction pass
// (bank-skewed) + 2 global atomics per channel — no serialized ds_bpermute.
// ---------------------------------------------------------------------------
__global__ __launch_bounds__(256)
void k_conv(const bf16_t* __restrict__ wfrag, const float* __restrict__ bias,
            const bf16_t* __restrict__ xt, bf16_t* __restrict__ traw,
            float* __restrict__ sum, float* __restrict__ sumsq) {
  extern __shared__ float lds_v[];                   // [256][128] f32 = 128 KB
  const int b = blockIdx.x / 18, pb = blockIdx.x % 18;
  const int tid = threadIdx.x;
  const int wv = tid >> 5, lane = tid & 31, l16 = lane & 15, half = lane >> 4;
  const int p0 = pb * 128 + wv * 16;
  const size_t xb = (size_t)b * NPB;

  // B fragments (x as K=c, N=p), contiguous from XT; reused across all o-tiles
  v16bf bfr[8];
  const bf16_t* brow = xt + xb + (size_t)(p0 + l16) * C + 16 * half;
#pragma unroll
  for (int kc = 0; kc < 8; ++kc) bfr[kc] = *(const v16bf*)(brow + kc * 32);

  for (int ot = 0; ot < 16; ++ot) {
    const bf16_t* wrow = wfrag + ((size_t)(ot * 8) * 32 + lane) * 16;
    v8f acc = {};
#pragma unroll
    for (int kc = 0; kc < 8; ++kc) {
      v16bf afr = *(const v16bf*)(wrow + (size_t)kc * 32 * 16);
      acc = __builtin_amdgcn_wmma_f32_16x16x32_bf16(false, afr, false, bfr[kc],
                                                    (short)0, acc, false, false);
    }
#pragma unroll
    for (int j = 0; j < 8; ++j) {
      const int o = ot * 16 + j + 8 * half;          // C/D layout: M = j + 8*(lane/16)
      const float v = acc[j] + bias[o];
      traw[xb + (size_t)o * HW + p0 + l16] = (bf16_t)v;
      lds_v[o * 128 + wv * 16 + l16] = v;            // stage for stats
    }
  }
  __syncthreads();
  // thread `o` reduces its channel over the 128 pixels this WG produced
  {
    const int o = tid;
    float s1 = 0.f, s2 = 0.f;
#pragma unroll 8
    for (int ii = 0; ii < 128; ++ii) {
      const float v = lds_v[o * 128 + ((ii + o) & 127)];   // bank-skewed
      s1 += v; s2 += v * v;
    }
    atomicAdd(&sum[o], s1);
    atomicAdd(&sumsq[o], s2);
  }
}

// ---------------------------------------------------------------------------
// Kernel 4: BN scale/shift from batch stats
// ---------------------------------------------------------------------------
__global__ void k_bnparams(const float* __restrict__ sum, const float* __restrict__ sumsq,
                           const float* __restrict__ gamma, const float* __restrict__ beta,
                           float* __restrict__ scale, float* __restrict__ shift) {
  const int o = threadIdx.x;
  const float mean = sum[o] * (1.0f / STAT_CNT);
  const float var  = sumsq[o] * (1.0f / STAT_CNT) - mean * mean;
  const float sc   = gamma[o] * rsqrtf(var + BN_EPS);
  scale[o] = sc;
  shift[o] = beta[o] - mean * sc;
}

// ---------------------------------------------------------------------------
// Kernel 5: apply BN+ReLU and transpose the flat-reinterpreted T (2304x256)
// into TT[k][m] (K-contiguous B-operand for matmul 2). Channel of flat index
// i = m*256+k is o = i/2304 = m/9 exactly (since (m%9)*256+k < 2304).
// ---------------------------------------------------------------------------
__global__ __launch_bounds__(256)
void k_transpose_t(const bf16_t* __restrict__ traw, const float* __restrict__ scale,
                   const float* __restrict__ shift, bf16_t* __restrict__ tt) {
  __shared__ float tile[32][33];
  const int m0 = blockIdx.x * 32, k0 = blockIdx.y * 32, b = blockIdx.z;
  const int tx = threadIdx.x, ty = threadIdx.y;
  const size_t xb = (size_t)b * NPB;
#pragma unroll
  for (int r = 0; r < 4; ++r) {
    const int m = m0 + ty + r * 8;
    const int o = m / 9;
    float f = (float)traw[xb + (size_t)m * C + k0 + tx];
    f = f * scale[o] + shift[o];
    tile[ty + r * 8][tx] = f > 0.f ? f : 0.f;
  }
  __syncthreads();
#pragma unroll
  for (int r = 0; r < 4; ++r)
    tt[xb + (size_t)(k0 + ty + r * 8) * HW + m0 + tx] = (bf16_t)tile[tx][ty + r * 8];
}

// ---------------------------------------------------------------------------
// Kernel 6: fused  Y = (relu(A1@B1 / n^2))^2 @ T  — S never leaves the WGP.
// 576 WGs = 32 batches x 18 row-blocks of 128; wave owns 16 rows x 256 cols.
// Per 32-wide m chunk: 16 WMMAs for S-tile, LDS layout fixup, 16 WMMAs for Y.
// ---------------------------------------------------------------------------
__global__ __launch_bounds__(256)
void k_main(const float* __restrict__ x, const bf16_t* __restrict__ xt,
            const bf16_t* __restrict__ tt, const float* __restrict__ inv2,
            float* __restrict__ out) {
  __shared__ bf16_t lds_s[8 * 16 * 32];              // per-wave 16x32 S tile
  const int b = blockIdx.x / 18, nblk = blockIdx.x % 18;
  const int tid = threadIdx.x;
  const int wv = tid >> 5, lane = tid & 31, l16 = lane & 15, half = lane >> 4;
  const int n0 = nblk * 128 + wv * 16;
  const size_t xb = (size_t)b * NPB;

  // Persistent A1 fragments: rows n0..n0+15, K=256 (8 chunks), f32 -> bf16.
  // One-time cost, amortized over 72 main-loop iterations.
  const float* xrow = x + xb + (size_t)(n0 + l16) * C;
  v16bf afr[8];
#pragma unroll
  for (int kc = 0; kc < 8; ++kc) {
    const int kbase = kc * 32 + 16 * half;
#pragma unroll
    for (int i = 0; i < 8; ++i) {
      afr[kc][i]     = (bf16_t)xrow[kbase + i];
      afr[kc][8 + i] = (bf16_t)xrow[kbase + 8 + i];
    }
  }
  const float iv = inv2[b];
  v8f yacc[16] = {};                                  // 16 N-tiles of Y (16x256 f32)

  bf16_t* sl = lds_s + wv * 16 * 32;
  const bf16_t* xtb = xt + xb;
  const bf16_t* ttb = tt + xb;

  for (int m0 = 0; m0 < HW; m0 += 32) {
    // ---- matmul 1: S rows [n0,n0+16) x cols [m0,m0+32), K=256
#pragma unroll
    for (int t = 0; t < 2; ++t) {
      v8f acc = {};
      const bf16_t* brow = xtb + (size_t)(m0 + t * 16 + l16) * C + 16 * half;
#pragma unroll
      for (int kc = 0; kc < 8; ++kc) {
        v16bf bfr = *(const v16bf*)(brow + kc * 32);
        acc = __builtin_amdgcn_wmma_f32_16x16x32_bf16(false, afr[kc], false, bfr,
                                                      (short)0, acc, false, false);
      }
      // s = (relu(cos) * inv_norm^2)^2 ; stash bf16 (keeps f32 exponent range;
      // fp16 would flush these ~1e-9 values to zero)
#pragma unroll
      for (int j = 0; j < 8; ++j) {
        const float v = acc[j] > 0.f ? acc[j] * iv : 0.f;
        sl[(j + 8 * half) * 32 + t * 16 + l16] = (bf16_t)(v * v);
      }
    }
    // re-read S tile in A-operand layout (same-wave DS ops are in-order)
    v16bf sfr;
    {
      const v8bf* p = (const v8bf*)(sl + l16 * 32);
      const v8bf lo = p[2 * half];                    // mm = 16*half .. +7
      const v8bf hi = p[2 * half + 1];                // mm = 16*half+8 .. +7
#pragma unroll
      for (int i = 0; i < 8; ++i) { sfr[i] = lo[i]; sfr[8 + i] = hi[i]; }
    }
    // ---- matmul 2: Y[16x256] += S(16x32) @ T(32x256)
    const bf16_t* trow = ttb + (size_t)l16 * HW + m0 + 16 * half;
    if (m0 + 32 < HW) {                               // hint next chunk into caches
      __builtin_prefetch(xtb + (size_t)(m0 + 32 + l16) * C, 0, 1);
      __builtin_prefetch(trow + 32, 0, 1);
    }
#pragma unroll
    for (int jt = 0; jt < 16; ++jt) {
      v16bf bfr = *(const v16bf*)(trow + (size_t)jt * 16 * HW);
      yacc[jt] = __builtin_amdgcn_wmma_f32_16x16x32_bf16(false, sfr, false, bfr,
                                                         (short)0, yacc[jt], false, false);
    }
  }
  // write Y (flat (b,hw,c) == output (b,c,h,w) memory)
  float* ob = out + xb;
#pragma unroll
  for (int jt = 0; jt < 16; ++jt)
#pragma unroll
    for (int j = 0; j < 8; ++j)
      ob[(size_t)(n0 + j + 8 * half) * C + jt * 16 + l16] = yacc[jt][j];
}

// ---------------------------------------------------------------------------
extern "C" void kernel_launch(void* const* d_in, const int* in_sizes, int n_in,
                              void* d_out, int out_size, void* d_ws, size_t ws_size,
                              hipStream_t stream) {
  const float* x      = (const float*)d_in[0];
  const float* conv_w = (const float*)d_in[1];
  const float* conv_b = (const float*)d_in[2];
  const float* gamma  = (const float*)d_in[3];
  const float* beta   = (const float*)d_in[4];
  float* out = (float*)d_out;

  char* ws = (char*)d_ws;
  const size_t BFB = (size_t)NB * NPB * sizeof(bf16_t);   // 37,748,736 B
  bf16_t* xt    = (bf16_t*)(ws);
  bf16_t* traw  = (bf16_t*)(ws + BFB);
  bf16_t* tt    = (bf16_t*)(ws + 2 * BFB);
  bf16_t* wfrag = (bf16_t*)(ws + 3 * BFB);                // 128 KB swizzled weights
  float*  inv2  = (float*)(ws + 3 * BFB + 131072);
  float*  sum   = inv2 + 32;
  float*  sumsq = sum + 256;
  float*  scale = sumsq + 256;
  float*  shift = scale + 256;

  hipMemsetAsync(sum, 0, 512 * sizeof(float), stream);    // zero BN accumulators

  k_norm<<<NB, 256, 0, stream>>>(x, inv2);
  k_transpose_x<<<dim3(HW / 32, C / 32, NB), dim3(32, 8), 0, stream>>>(x, xt);
  k_wcast<<<256, 256, 0, stream>>>(conv_w, wfrag);
  k_conv<<<NB * 18, 256, 131072, stream>>>(wfrag, conv_b, xt, traw, sum, sumsq);
  k_bnparams<<<1, 256, 0, stream>>>(sum, sumsq, gamma, beta, scale, shift);
  k_transpose_t<<<dim3(HW / 32, C / 32, NB), dim3(32, 8), 0, stream>>>(traw, scale, shift, tt);
  k_main<<<NB * 18, 256, 0, stream>>>(x, xt, tt, inv2, out);
}
